// EntityLabeler_54271206752376
// MI455X (gfx1250) — compile-verified, matrix operand order checked by reading, not run on hardware
//
#include <hip/hip_runtime.h>
#include <hip/hip_bf16.h>
#include <math.h>

// ---------------- problem constants ----------------
#define BB   128
#define SS   512
#define VV   32000
#define EE   256
#define HH   256
#define GG   1024      // 4*H
#define LL   9
#define NBLK 32        // persistent LSTM blocks (32 blk x 256 thr = 256 waves = 8192 threads)

typedef __attribute__((ext_vector_type(16))) __bf16 v16bf;
typedef __attribute__((ext_vector_type(8)))  __bf16 v8bf;
typedef __attribute__((ext_vector_type(8)))  float  v8f;
typedef __attribute__((ext_vector_type(4)))  int    v4i;

// pointer-to-v4i in global (addrspace 1) / LDS (addrspace 3) address spaces,
// matching the async-to-LDS builtin's parameter types.
typedef __attribute__((address_space(1))) v4i* gv4i_p;
typedef __attribute__((address_space(3))) v4i* lv4i_p;

#if __has_builtin(__builtin_amdgcn_global_load_async_to_lds_b128)
#define HAVE_ASYNC_LDS 1
#else
#define HAVE_ASYNC_LDS 0
#endif

// ---------------- WMMA fragment helpers ----------------
// 16-bit A (16x32, MxK) / B (32x16, KxN) fragment layout (ISA 7.12.2):
// lanes 0-15 : row/col = lane,    K = [kb+0..kb+7]  then [kb+16..kb+23]
// lanes 16-31: row/col = lane-16, K = [kb+8..kb+15] then [kb+24..kb+31]
// With K-contiguous bf16 storage this is exactly two b128 loads per lane.
__device__ __forceinline__ v16bf load_frag_kcontig(const __bf16* __restrict__ base,
                                                   int rowcol, int ld, int kb, int half) {
  const __bf16* p = base + (size_t)rowcol * ld + kb + (half << 3);
  v8bf lo = *(const v8bf*)(p);
  v8bf hi = *(const v8bf*)(p + 16);
  v16bf r;
#pragma unroll
  for (int i = 0; i < 8; ++i) { r[i] = lo[i]; r[8 + i] = hi[i]; }
  return r;
}

__device__ __forceinline__ float sigf(float x) { return 1.0f / (1.0f + expf(-x)); }

// ---------------- grid-wide two-phase barrier ----------------
__device__ __forceinline__ void grid_barrier(unsigned* cnt, unsigned* gen, unsigned nblocks) {
  __threadfence();          // make this thread's global writes visible device-wide
  __syncthreads();
  if (threadIdx.x == 0) {
    unsigned g = __hip_atomic_load(gen, __ATOMIC_ACQUIRE, __HIP_MEMORY_SCOPE_AGENT);
    unsigned arrived = __hip_atomic_fetch_add(cnt, 1u, __ATOMIC_ACQ_REL, __HIP_MEMORY_SCOPE_AGENT);
    if (arrived == nblocks - 1u) {
      __hip_atomic_store(cnt, 0u, __ATOMIC_RELAXED, __HIP_MEMORY_SCOPE_AGENT);
      __hip_atomic_fetch_add(gen, 1u, __ATOMIC_ACQ_REL, __HIP_MEMORY_SCOPE_AGENT);
    } else {
      while (__hip_atomic_load(gen, __ATOMIC_ACQUIRE, __HIP_MEMORY_SCOPE_AGENT) == g) {
        __builtin_amdgcn_s_sleep(2);
      }
    }
  }
  __syncthreads();
}

// ---------------- fp32 -> bf16 conversion ----------------
__global__ void __launch_bounds__(256) cvt_bf16_kernel(const float* __restrict__ in,
                                                       __bf16* __restrict__ out, int n) {
  int i = blockIdx.x * 256 + threadIdx.x;
  if (i < n) out[i] = (__bf16)in[i];
}

// ---------------- xp = gather(emb)[.] @ W_ih^T + b_ih ----------------
// Output xp[(s*B+b)][g] fp32, i.e. einsum('bse,ge->sbg').
// grid = (M/128, N/64) = (512, 16), block = 256 (8 waves), wave = one 16x64 strip.
__global__ void __launch_bounds__(256) xp_gemm_kernel(
    const int*    __restrict__ src,    // [B][S]
    const __bf16* __restrict__ embb,   // [V][E] bf16
    const __bf16* __restrict__ wih,    // [4H][E] bf16 (row g, K-contiguous)
    const float*  __restrict__ b_ih,   // [4H]
    float*        __restrict__ xp)     // [S*B][4H]
{
  const int lane  = threadIdx.x & 31;
  const int wave  = threadIdx.x >> 5;
  const int ln15  = lane & 15;
  const int half  = lane >> 4;
  const int m_base = (blockIdx.x * 8 + wave) * 16;
  const int n_base = blockIdx.y * 64;

  // token row for this lane's A-fragment row (fixed across K)
  const int row   = m_base + ln15;          // row = s*B + b
  const int b     = row & (BB - 1);
  const int s     = row >> 7;
  const int token = src[b * SS + s];
  const __bf16* arow = embb + (size_t)token * EE;

  v8f acc[4];
#pragma unroll
  for (int t = 0; t < 4; ++t) {
    float bias = b_ih[n_base + t * 16 + ln15];
#pragma unroll
    for (int r = 0; r < 8; ++r) acc[t][r] = bias;
  }

#pragma unroll
  for (int kk = 0; kk < EE / 32; ++kk) {
    const int kb = kk * 32;
    v16bf afrag = load_frag_kcontig(arow, 0, 0, kb, half);
#pragma unroll
    for (int t = 0; t < 4; ++t) {
      v16bf bfrag = load_frag_kcontig(wih, n_base + t * 16 + ln15, EE, kb, half);
      acc[t] = __builtin_amdgcn_wmma_f32_16x16x32_bf16(false, afrag, false, bfrag,
                                                       (short)0, acc[t], false, false);
    }
  }

  // C/D layout: VGPR r -> M = r + 8*half, N = ln15
#pragma unroll
  for (int t = 0; t < 4; ++t) {
    const int col = n_base + t * 16 + ln15;
#pragma unroll
    for (int r = 0; r < 8; ++r) {
      const int rr = m_base + r + (half << 3);
      xp[(size_t)rr * GG + col] = acc[t][r];
    }
  }
}

// ---------------- persistent LSTM recurrence ----------------
// 32 blocks x 256 threads (8 waves). Block owns one 16-row M tile and a 256-col
// gate strip: tiles [blockIdx*16, +16), all sharing m_base = (blockIdx/4)*16.
// W_hh B-fragments are step-invariant per wave -> held in registers (128 VGPRs).
// The 8KB h-tile is staged into LDS each step (async-to-LDS when available).
__global__ void __launch_bounds__(256, 1) lstm_persistent_kernel(
    const float*  __restrict__ xp,     // [S][B][4H] fp32 (includes b_ih)
    const __bf16* __restrict__ whh,    // [4H][H] bf16 (row g, K-contiguous)
    const float*  __restrict__ b_hh,   // [4H]
    __bf16*       __restrict__ hbf,    // [B][H] bf16 running hidden state
    float*        __restrict__ cst,    // [B][H] fp32 cell state
    float*        __restrict__ gates,  // [B][4H] fp32 scratch
    float*        __restrict__ hall,   // [S][B][H] fp32 (pre-relu hidden states)
    unsigned*     __restrict__ bar)    // [0]=count, [1]=generation
{
  __shared__ __bf16 hs[16 * HH];      // 8 KiB: this block's 16 h rows

  const int lane  = threadIdx.x & 31;
  const int ln15  = lane & 15;
  const int half  = lane >> 4;
  const int wave  = threadIdx.x >> 5;
  const int tid   = blockIdx.x * 256 + threadIdx.x;   // 0..8191

  const int m_base = (blockIdx.x >> 2) * 16;          // block-constant M tile
  const int col0   = ((blockIdx.x & 3) * 16 + wave * 2 + 0) * 16 + ln15;
  const int col1   = col0 + 16;
  const float bias0 = b_hh[col0];
  const float bias1 = b_hh[col1];

  // step-invariant W_hh^T B-fragments, resident in VGPRs for all 512 steps
  v16bf wfrag[2][8];
#pragma unroll
  for (int kk = 0; kk < HH / 32; ++kk) {
    wfrag[0][kk] = load_frag_kcontig(whh, col0, HH, kk * 32, half);
    wfrag[1][kk] = load_frag_kcontig(whh, col1, HH, kk * 32, half);
  }

  const __bf16* __restrict__ hsrc = hbf + (size_t)m_base * HH;  // 16 contiguous rows
  const __bf16* hs_c = hs;

  for (int s = 0; s < SS; ++s) {
    const float* xps = xp + (size_t)s * (BB * GG);

    // ---- stage this block's 16 h rows (8 KiB) into LDS ----
#if HAVE_ASYNC_LDS
#pragma unroll
    for (int c = 0; c < 2; ++c) {
      const int chunk = threadIdx.x + c * 256;          // 512 x 16B chunks
      __builtin_amdgcn_global_load_async_to_lds_b128(
          (gv4i_p)(hsrc + chunk * 8),
          (lv4i_p)(hs + chunk * 8), 0, 0);
    }
#if __has_builtin(__builtin_amdgcn_s_wait_asynccnt)
    __builtin_amdgcn_s_wait_asynccnt(0);
#else
    asm volatile("s_wait_asynccnt 0x0" ::: "memory");
#endif
    __syncthreads();
#else
#pragma unroll
    for (int c = 0; c < 2; ++c) {
      const int chunk = threadIdx.x + c * 256;
      *(v8bf*)(hs + chunk * 8) = *(const v8bf*)(hsrc + chunk * 8);
    }
    __syncthreads();
#endif

    // ---- phase 1: gates = xp[s] + h @ W_hh^T + b_hh ----
    v16bf afrag[HH / 32];
#pragma unroll
    for (int kk = 0; kk < HH / 32; ++kk)
      afrag[kk] = load_frag_kcontig(hs_c, ln15, HH, kk * 32, half);

#pragma unroll
    for (int tt = 0; tt < 2; ++tt) {
      const int col    = tt ? col1 : col0;
      const float bias = tt ? bias1 : bias0;
      v8f acc;
#pragma unroll
      for (int r = 0; r < 8; ++r)
        acc[r] = xps[(size_t)(m_base + r + (half << 3)) * GG + col] + bias;
#pragma unroll
      for (int kk = 0; kk < HH / 32; ++kk)
        acc = __builtin_amdgcn_wmma_f32_16x16x32_bf16(false, afrag[kk], false, wfrag[tt][kk],
                                                      (short)0, acc, false, false);
#pragma unroll
      for (int r = 0; r < 8; ++r)
        gates[(size_t)(m_base + r + (half << 3)) * GG + col] = acc[r];
    }

    // hint next step's xp slab toward the caches (global_prefetch_b8)
    if (s + 1 < SS)
      __builtin_prefetch(xp + (size_t)(s + 1) * (BB * GG) + (size_t)tid * 16, 0, 1);

    grid_barrier(&bar[0], &bar[1], NBLK);

    // ---- phase 2: cell update ----
#pragma unroll
    for (int i = 0; i < 4; ++i) {
      const int e = tid + i * (NBLK * 256);   // 0..32767
      const int b = e >> 8;
      const int j = e & (HH - 1);
      const float* g = gates + (size_t)b * GG;
      const float gi = g[j];
      const float gf = g[j + HH];
      const float gg = g[j + 2 * HH];
      const float go = g[j + 3 * HH];
      const float c0 = cst[e];
      const float cn = sigf(gf) * c0 + sigf(gi) * tanhf(gg);
      const float hn = sigf(go) * tanhf(cn);
      cst[e]  = cn;
      hbf[e]  = (__bf16)hn;
      hall[(size_t)s * (BB * HH) + e] = hn;
    }

    grid_barrier(&bar[0], &bar[1], NBLK);
  }
}

// ---------------- emissions = relu(h) @ W_lin^T + b_lin ----------------
// one wave per (s,b) pair; wave32 shuffle reduction over H=256.
__global__ void __launch_bounds__(256) emis_kernel(
    const float* __restrict__ hall,   // [S*B][H]
    const float* __restrict__ wlin,   // [L][H]
    const float* __restrict__ blin,   // [L]
    float*       __restrict__ em)     // [B][S][L]
{
  const int lane = threadIdx.x & 31;
  const int p    = (blockIdx.x * 256 + threadIdx.x) >> 5;  // 0..65535 = s*B+b
  const int s    = p >> 7;
  const int b    = p & (BB - 1);

  const float* hr = hall + (size_t)p * HH + lane * 8;
  float hv[8];
#pragma unroll
  for (int i = 0; i < 8; ++i) hv[i] = fmaxf(hr[i], 0.0f);

  float* eo = em + ((size_t)b * SS + s) * LL;
#pragma unroll
  for (int l = 0; l < LL; ++l) {
    const float* w = wlin + l * HH + lane * 8;
    float partial = 0.0f;
#pragma unroll
    for (int i = 0; i < 8; ++i) partial += hv[i] * w[i];
#pragma unroll
    for (int off = 16; off >= 1; off >>= 1) partial += __shfl_down(partial, off);
    if (lane == 0) eo[l] = partial + blin[l];
  }
}

// ---------------- CRF negative log-likelihood ----------------
// one wave per batch row; 9-tag alpha in LDS; atomicAdd scalar result.
__global__ void __launch_bounds__(32) crf_kernel(
    const float* __restrict__ em,       // [B][S][L]
    const int*   __restrict__ labels,   // [B][S]
    const unsigned char* __restrict__ masks, // [B][S] (bool)
    const float* __restrict__ start_t,  // [L]
    const float* __restrict__ end_t,    // [L]
    const float* __restrict__ trans,    // [L][L]
    float*       __restrict__ out)      // scalar
{
  const int b    = blockIdx.x;
  const int lane = threadIdx.x;
  __shared__ float alpha[16];
  __shared__ float tsh[LL * LL];

  const float* e = em + (size_t)b * SS * LL;
  const int* lab = labels + (size_t)b * SS;
  const unsigned char* mk = masks + (size_t)b * SS;

  for (int i = lane; i < LL * LL; i += 32) tsh[i] = trans[i];
  if (lane < LL) alpha[lane] = start_t[lane] + e[lane];
  __syncthreads();

  for (int s = 1; s < SS; ++s) {
    float anew = 0.0f;
    if (lane < LL) {
      const float aold = alpha[lane];
      const float es   = e[s * LL + lane];
      float v[LL];
      float mx = -3.4e38f;
#pragma unroll
      for (int l = 0; l < LL; ++l) {
        v[l] = alpha[l] + tsh[l * LL + lane] + es;
        mx = fmaxf(mx, v[l]);
      }
      float sum = 0.0f;
#pragma unroll
      for (int l = 0; l < LL; ++l) sum += expf(v[l] - mx);
      const float nxt = mx + logf(sum);
      anew = mk[s] ? nxt : aold;
    }
    __syncthreads();
    if (lane < LL) alpha[lane] = anew;
    __syncthreads();
  }

  if (lane == 0) {
    float mx = -3.4e38f;
    for (int l = 0; l < LL; ++l) mx = fmaxf(mx, alpha[l] + end_t[l]);
    float sum = 0.0f;
    for (int l = 0; l < LL; ++l) sum += expf(alpha[l] + end_t[l] - mx);
    const float logZ = mx + logf(sum);

    float score = start_t[lab[0]] + e[lab[0]];
    int cnt = mk[0] ? 1 : 0;
    for (int s = 1; s < SS; ++s) {
      const float mf = mk[s] ? 1.0f : 0.0f;
      score += (tsh[lab[s - 1] * LL + lab[s]] + e[s * LL + lab[s]]) * mf;
      cnt   += mk[s] ? 1 : 0;
    }
    score += end_t[lab[cnt - 1]];
    atomicAdd(out, logZ - score);
  }
}

// ---------------- workspace layout (bytes, all 256-aligned) ----------------
static constexpr size_t SZ_XP    = (size_t)SS * BB * GG * 4;   // 256 MiB
static constexpr size_t SZ_HALL  = (size_t)SS * BB * HH * 4;   //  64 MiB
static constexpr size_t SZ_EMBB  = (size_t)VV * EE * 2;        //  ~15.6 MiB
static constexpr size_t SZ_WIH   = (size_t)GG * EE * 2;        //  512 KiB
static constexpr size_t SZ_WHH   = (size_t)GG * HH * 2;        //  512 KiB
static constexpr size_t SZ_GATES = (size_t)BB * GG * 4;        //  512 KiB
static constexpr size_t SZ_HBF   = (size_t)BB * HH * 2;        //   64 KiB
static constexpr size_t SZ_CST   = (size_t)BB * HH * 4;        //  128 KiB
static constexpr size_t SZ_EM    = (size_t)BB * SS * LL * 4;   // ~2.25 MiB

static constexpr size_t OFF_XP    = 0;
static constexpr size_t OFF_HALL  = OFF_XP    + SZ_XP;
static constexpr size_t OFF_EMBB  = OFF_HALL  + SZ_HALL;
static constexpr size_t OFF_WIH   = OFF_EMBB  + SZ_EMBB;
static constexpr size_t OFF_WHH   = OFF_WIH   + SZ_WIH;
static constexpr size_t OFF_GATES = OFF_WHH   + SZ_WHH;
static constexpr size_t OFF_HBF   = OFF_GATES + SZ_GATES;
static constexpr size_t OFF_CST   = OFF_HBF   + SZ_HBF;
static constexpr size_t OFF_EM    = OFF_CST   + SZ_CST;
static constexpr size_t OFF_BAR   = OFF_EM    + SZ_EM;

extern "C" void kernel_launch(void* const* d_in, const int* in_sizes, int n_in,
                              void* d_out, int out_size, void* d_ws, size_t ws_size,
                              hipStream_t stream) {
  (void)in_sizes; (void)n_in; (void)out_size; (void)ws_size;

  const int*   src     = (const int*)  d_in[0];
  const int*   labels  = (const int*)  d_in[1];
  const unsigned char* masks = (const unsigned char*)d_in[2];
  const float* emb     = (const float*)d_in[3];
  const float* w_ih    = (const float*)d_in[4];
  const float* w_hh    = (const float*)d_in[5];
  const float* b_ih    = (const float*)d_in[6];
  const float* b_hh    = (const float*)d_in[7];
  const float* w_lin   = (const float*)d_in[8];
  const float* b_lin   = (const float*)d_in[9];
  const float* start_t = (const float*)d_in[10];
  const float* end_t   = (const float*)d_in[11];
  const float* trans   = (const float*)d_in[12];

  char* ws = (char*)d_ws;
  float*    xp    = (float*)   (ws + OFF_XP);
  float*    hall  = (float*)   (ws + OFF_HALL);
  __bf16*   embb  = (__bf16*)  (ws + OFF_EMBB);
  __bf16*   wihb  = (__bf16*)  (ws + OFF_WIH);
  __bf16*   whhb  = (__bf16*)  (ws + OFF_WHH);
  float*    gates = (float*)   (ws + OFF_GATES);
  __bf16*   hbf   = (__bf16*)  (ws + OFF_HBF);
  float*    cst   = (float*)   (ws + OFF_CST);
  float*    em    = (float*)   (ws + OFF_EM);
  unsigned* bar   = (unsigned*)(ws + OFF_BAR);
  float*    out   = (float*)d_out;

  // zero-init states / barrier / output (graph-capture-safe)
  (void)hipMemsetAsync(hbf, 0, SZ_HBF, stream);
  (void)hipMemsetAsync(cst, 0, SZ_CST, stream);
  (void)hipMemsetAsync(bar, 0, 2 * sizeof(unsigned), stream);
  (void)hipMemsetAsync(out, 0, sizeof(float), stream);

  // fp32 -> bf16 weight/table conversion
  cvt_bf16_kernel<<<(VV * EE + 255) / 256, 256, 0, stream>>>(emb, embb, VV * EE);
  cvt_bf16_kernel<<<(GG * EE + 255) / 256, 256, 0, stream>>>(w_ih, wihb, GG * EE);
  cvt_bf16_kernel<<<(GG * HH + 255) / 256, 256, 0, stream>>>(w_hh, whhb, GG * HH);

  // xp = embed-gather GEMM  (M=65536, N=1024, K=256)
  dim3 ggrid(SS * BB / 128, GG / 64);
  xp_gemm_kernel<<<ggrid, 256, 0, stream>>>(src, embb, wihb, b_ih, xp);

  // persistent LSTM recurrence over 512 steps
  lstm_persistent_kernel<<<NBLK, 256, 0, stream>>>(xp, whhb, b_hh, hbf, cst, gates, hall, bar);

  // emissions
  emis_kernel<<<SS * BB / 8, 256, 0, stream>>>(hall, w_lin, b_lin, em);

  // CRF NLL
  crf_kernel<<<BB, 32, 0, stream>>>(em, labels, masks, start_t, end_t, trans, out);
}
